// Attention_32409823216264
// MI455X (gfx1250) — matrix operand order repaired and checked by reading, not compiled, not run on hardware
//
#include <hip/hip_runtime.h>
#include <stdint.h>

// Attention fwd: B=2,H=16,S=2048,D=64 on gfx1250 (wave32 WMMA).
// d_out = p_val [B,H,S,D] ++ p_attn [B,H,S,S] (f32).
//
// Roofline: mandatory HBM traffic ~600MB (536MB p_attn write) => ~26us floor
// @23.3TB/s. Matmul work (3x QK recompute + PV = ~69 GFLOP) runs on the
// matrix pipe (co-executes with VALU), so the design goal is minimal
// per-element VALU work: 3-pass shift-invariant softmax with
//   pass1: per-lane running max only (1 fmax/elt, no mask, no shuffles)
//   pass2: ev=exp(s-m), per-lane l+=ev, unnormalized O+=ev*V via WMMA
//   pass3: p=exp(s-m)/l -> the single big p_attn write
// Cross-lane reductions deferred to one 4-step xor-merge per pass.

typedef __attribute__((ext_vector_type(16))) _Float16 v16h;
typedef __attribute__((ext_vector_type(8)))  _Float16 v8h;
typedef __attribute__((ext_vector_type(4)))  _Float16 v4h;
typedef __attribute__((ext_vector_type(8)))  float    v8f;

#define Sn 2048
#define Dn 64
#define KT 32      // k-chunk (one WMMA contraction for P*V)
#define QT 64      // q rows per block (4 waves x 16)
#define KPITCH 72  // Qs/Ks row pitch (f16): 16B multiple, 16 distinct banks for 16-lane gathers
#define VPITCH 40  // Vt/Ps row pitch (f16): 16B multiple, conflict-free b128 gathers

union V16HU { v16h v; v8h h[2]; };
__device__ __forceinline__ v16h ld16_2(const _Float16* p0, const _Float16* p1) {
    V16HU u; u.h[0] = *(const v8h*)p0; u.h[1] = *(const v8h*)p1; return u.v;
}
__device__ __forceinline__ v16h ld16_c(const _Float16* p) { return ld16_2(p, p + 8); }

#define WMMA_F16(a, b, c) \
    __builtin_amdgcn_wmma_f32_16x16x32_f16(false, (a), false, (b), (short)0, (c), false, false)

__global__ void __launch_bounds__(128)
attn_fwd_kernel(const float* __restrict__ Qg, const float* __restrict__ Kg,
                const float* __restrict__ Vg, const unsigned char* __restrict__ Mg,
                float* __restrict__ outV, float* __restrict__ outA)
{
    __shared__ __align__(16) _Float16 Qs[QT][KPITCH];    // 9.0 KB
    __shared__ __align__(16) _Float16 Ks[KT][KPITCH];    // 4.5 KB
    __shared__ __align__(16) _Float16 Vt[Dn][VPITCH];    // 5.0 KB (V transposed [d][k])
    __shared__ __align__(16) _Float16 Ps[4][16][VPITCH]; // 5.0 KB (per-wave ev tile)

    const int tid  = threadIdx.x;
    const int wv   = tid >> 5;
    const int lane = tid & 31;
    const int h2   = lane >> 4;
    const int ln   = lane & 15;

    const int bh = blockIdx.x >> 5;           // b*H + h
    const int q0 = (blockIdx.x & 31) * QT;
    const size_t qkBase = (size_t)bh * Sn * Dn;

    // ---- stage Q tile (f32 -> f16), vectorized ----
    for (int i = tid; i < (QT * Dn) / 4; i += 128) {
        const int r = i >> 4, d4 = (i & 15) << 2;
        const float4 f = *(const float4*)(Qg + qkBase + (size_t)(q0 + r) * Dn + d4);
        v4h h = {(_Float16)f.x, (_Float16)f.y, (_Float16)f.z, (_Float16)f.w};
        *(v4h*)&Qs[r][d4] = h;
    }
    __syncthreads();

    // A-operand gather for Q (contiguous k = 16*h2 + s; B gathers use the same order)
    const int qr = wv * 16 + ln;
    const v16h qa0 = ld16_c(&Qs[qr][16 * h2]);       // d 0..31
    const v16h qa1 = ld16_c(&Qs[qr][32 + 16 * h2]);  // d 32..63

    const int qgBase = q0 + wv * 16 + 8 * h2;   // + r => global q row of C element r

    // =================== pass 1: per-lane running max (mask-free) ===================
    float m8[8];
#pragma unroll
    for (int r = 0; r < 8; ++r) m8[r] = -3.0e38f;

    for (int kb = 0; kb < Sn; kb += KT) {
        __syncthreads();
        for (int i = tid; i < (KT * Dn) / 4; i += 128) {
            const int n = i >> 4, d4 = (i & 15) << 2;
            const float4 f = *(const float4*)(Kg + qkBase + (size_t)(kb + n) * Dn + d4);
            v4h h = {(_Float16)f.x, (_Float16)f.y, (_Float16)f.z, (_Float16)f.w};
            *(v4h*)&Ks[n][d4] = h;
        }
        __syncthreads();
#pragma unroll
        for (int sub = 0; sub < 2; ++sub) {
            const int n0 = sub * 16;
            v16h b0 = ld16_c(&Ks[n0 + ln][16 * h2]);
            v16h b1 = ld16_c(&Ks[n0 + ln][32 + 16 * h2]);
            v8f c = {};
            c = WMMA_F16(qa0, b0, c);
            c = WMMA_F16(qa1, b1, c);
#pragma unroll
            for (int r = 0; r < 8; ++r) m8[r] = fmaxf(m8[r], c[r]);
        }
    }
    // one deferred cross-lane max reduction; fold in the 1/sqrt(D) scale
#pragma unroll
    for (int r = 0; r < 8; ++r) {
#pragma unroll
        for (int o = 8; o >= 1; o >>= 1) m8[r] = fmaxf(m8[r], __shfl_xor(m8[r], o, 32));
        m8[r] *= 0.125f;
    }

    // =================== pass 2: l = sum(exp), unnormalized O += ev*V ===================
    float l8[8];
#pragma unroll
    for (int r = 0; r < 8; ++r) l8[r] = 0.0f;
    v8f acc0 = {}, acc1 = {}, acc2 = {}, acc3 = {};

    for (int kb = 0; kb < Sn; kb += KT) {
        __syncthreads();
        for (int i = tid; i < (KT * Dn) / 4; i += 128) {
            const int n = i >> 4, d4 = (i & 15) << 2;
            const size_t g = qkBase + (size_t)(kb + n) * Dn + d4;
            const float4 fk = *(const float4*)(Kg + g);
            v4h hk = {(_Float16)fk.x, (_Float16)fk.y, (_Float16)fk.z, (_Float16)fk.w};
            *(v4h*)&Ks[n][d4] = hk;
            const float4 fv = *(const float4*)(Vg + g);
            Vt[d4 + 0][n] = (_Float16)fv.x;    // transpose so B gathers are contiguous
            Vt[d4 + 1][n] = (_Float16)fv.y;
            Vt[d4 + 2][n] = (_Float16)fv.z;
            Vt[d4 + 3][n] = (_Float16)fv.w;
        }
        __syncthreads();
#pragma unroll
        for (int sub = 0; sub < 2; ++sub) {
            const int n0 = sub * 16;
            v16h b0 = ld16_c(&Ks[n0 + ln][16 * h2]);
            v16h b1 = ld16_c(&Ks[n0 + ln][32 + 16 * h2]);
            v8f c = {};
            c = WMMA_F16(qa0, b0, c);
            c = WMMA_F16(qa1, b1, c);
            const int kg = kb + n0 + ln;
#pragma unroll
            for (int r = 0; r < 8; ++r) {
                const int qg = qgBase + r;
                const float s  = Mg[(size_t)qg * Sn + kg] ? -1.0e9f : c[r] * 0.125f;
                const float ev = __expf(s - m8[r]);   // in (0,1]: f16-safe
                l8[r] += ev;
                Ps[wv][r + 8 * h2][n0 + ln] = (_Float16)ev;
            }
        }
        __syncthreads();
        v16h pa = ld16_c(&Ps[wv][ln][16 * h2]);
        acc0 = WMMA_F16(pa, ld16_c(&Vt[ 0 + ln][16 * h2]), acc0);
        acc1 = WMMA_F16(pa, ld16_c(&Vt[16 + ln][16 * h2]), acc1);
        acc2 = WMMA_F16(pa, ld16_c(&Vt[32 + ln][16 * h2]), acc2);
        acc3 = WMMA_F16(pa, ld16_c(&Vt[48 + ln][16 * h2]), acc3);
    }
    // one deferred cross-lane sum reduction
    float il8[8];
#pragma unroll
    for (int r = 0; r < 8; ++r) {
#pragma unroll
        for (int o = 8; o >= 1; o >>= 1) l8[r] += __shfl_xor(l8[r], o, 32);
        il8[r] = 1.0f / fmaxf(l8[r], 1e-30f);
    }

    // ---- write p_val = O * (1/l) ----
#pragma unroll
    for (int r = 0; r < 8; ++r) {
        const int qg = qgBase + r;
        const size_t base = ((size_t)bh * Sn + qg) * Dn;
        outV[base +  0 + ln] = acc0[r] * il8[r];
        outV[base + 16 + ln] = acc1[r] * il8[r];
        outV[base + 32 + ln] = acc2[r] * il8[r];
        outV[base + 48 + ln] = acc3[r] * il8[r];
    }

    // =================== pass 3: p_attn = exp(s-m)/l (the 536MB write) ===================
    for (int kb = 0; kb < Sn; kb += KT) {
        __syncthreads();
        for (int i = tid; i < (KT * Dn) / 4; i += 128) {
            const int n = i >> 4, d4 = (i & 15) << 2;
            const float4 f = *(const float4*)(Kg + qkBase + (size_t)(kb + n) * Dn + d4);
            v4h h = {(_Float16)f.x, (_Float16)f.y, (_Float16)f.z, (_Float16)f.w};
            *(v4h*)&Ks[n][d4] = h;
        }
        __syncthreads();
#pragma unroll
        for (int sub = 0; sub < 2; ++sub) {
            const int n0 = sub * 16;
            v16h b0 = ld16_c(&Ks[n0 + ln][16 * h2]);
            v16h b1 = ld16_c(&Ks[n0 + ln][32 + 16 * h2]);
            v8f c = {};
            c = WMMA_F16(qa0, b0, c);
            c = WMMA_F16(qa1, b1, c);
            const int kg = kb + n0 + ln;
#pragma unroll
            for (int r = 0; r < 8; ++r) {
                const int qg = qgBase + r;
                const float s = Mg[(size_t)qg * Sn + kg] ? -1.0e9f : c[r] * 0.125f;
                outA[((size_t)bh * Sn + qg) * Sn + kg] = __expf(s - m8[r]) * il8[r];
            }
        }
    }
}

extern "C" void kernel_launch(void* const* d_in, const int* in_sizes, int n_in,
                              void* d_out, int out_size, void* d_ws, size_t ws_size,
                              hipStream_t stream) {
    (void)in_sizes; (void)n_in; (void)out_size; (void)d_ws; (void)ws_size;
    const float* Q = (const float*)d_in[0];
    const float* K = (const float*)d_in[1];
    const float* V = (const float*)d_in[2];
    const unsigned char* M = (const unsigned char*)d_in[3];  // jnp.bool_ = 1 byte

    float* outV = (float*)d_out;                        // p_val: B*H*S*D
    float* outA = outV + (size_t)2 * 16 * Sn * Dn;      // p_attn: B*H*S*S

    const int blocks = 2 * 16 * (Sn / QT);  // 1024
    attn_fwd_kernel<<<blocks, 128, 0, stream>>>(Q, K, V, M, outV, outA);
}
